// TrajEstimation_88424786690274
// MI455X (gfx1250) — matrix operand order repaired and checked by reading, not compiled, not run on hardware
//
#include <hip/hip_runtime.h>
#include <stdint.h>

// ---------------------------------------------------------------------------
// CDNA5 / gfx1250 persistent-transformer rollout.
//   - 6 workgroups (one per mode), 512 threads = 16 wave32 waves each.
//   - bf16 WMMA (v_wmma_f32_16x16x32_bf16) for ctx@W_o, FF1, FF2,
//     register-blocked 2x2 tiles/wave: 4 independent accumulators, each
//     A/B fragment reused twice -> 2 b128 loads per WMMA.
//   - qkv GEMM algebraically folded (feat/extras constant parts precomputed,
//     K/V never materialized; softmax constant term cancels).
//   - fus@W_fus@W_out collapsed into a precomputed 967x2 matrix.
//   - scalar-stage weights stored transposed (contiguous inner reads).
// ---------------------------------------------------------------------------

typedef __attribute__((ext_vector_type(16))) __bf16 v16bf;
typedef __attribute__((ext_vector_type(8)))  float  v8f;

#define DEV __device__ __forceinline__

constexpr int B_    = 256;
constexpr int MODES = 6;
constexpr int INC_  = 448;
constexpr int EMB_  = 32;
constexpr int NH_   = 8;
constexpr int DFF_  = 2048;
constexpr int OBS_  = 20;
constexpr int HOR_  = 30;
constexpr int D_    = 512;
constexpr int FUS_  = 967;   // D + 7 + INC
constexpr int BLOCK = 512;   // 16 waves of 32

constexpr size_t aln(size_t x) { return (x + 255) & ~size_t(255); }

// ---- workspace layout (shared precompute) ----
constexpr size_t OFF_WOT    = 0;                                              // [512][512]  bf16 (W_o^T)
constexpr size_t OFF_WFF1T  = aln(OFF_WOT    + size_t(D_)  * D_   * 2);       // [2048][512] bf16 (W_ff1^T)
constexpr size_t OFF_WFF2T  = aln(OFF_WFF1T  + size_t(DFF_) * D_  * 2);       // [512][2048] bf16 (W_ff2^T)
constexpr size_t OFF_FEATC  = aln(OFF_WFF2T  + size_t(D_)  * DFF_ * 2);       // [256][1536] f32 feat@Wqkv[64:]+b
constexpr size_t OFF_WEXT   = aln(OFF_FEATC  + size_t(B_)  * 1536 * 4);       // [1536][8]   f32 reduced extras rows (T)
constexpr size_t OFF_WQT    = aln(OFF_WEXT   + size_t(1536) * 8 * 4);         // [512][32]   f32 Wq emb block (T)
constexpr size_t OFF_WVT    = aln(OFF_WQT    + size_t(D_) * EMB_ * 4);        // [512][32]   f32 Wv emb block (T)
constexpr size_t OFF_EMBOBS = aln(OFF_WVT    + size_t(D_) * EMB_ * 4);        // [20][256][32] f32
constexpr size_t OFF_WFG    = aln(OFF_EMBOBS + size_t(OBS_) * B_ * EMB_ * 4); // [967][2] f32 W_fus@W_out
constexpr size_t OFF_BFG    = aln(OFF_WFG    + size_t(FUS_) * 2 * 4);         // [2] f32
constexpr size_t OFF_FEATFG = aln(OFF_BFG    + 2 * 4);                        // [256][2] f32
constexpr size_t OFF_MODE   = aln(OFF_FEATFG + size_t(B_) * 2 * 4);

// ---- per-mode layout ----
constexpr size_t MO_EMB30  = 0;                                               // [30][256][32] f32
constexpr size_t MO_Q      = aln(MO_EMB30  + size_t(HOR_) * B_ * EMB_ * 4);   // [256][512] f32 (q, reused as h2)
constexpr size_t MO_EXV    = aln(MO_Q      + size_t(B_) * D_ * 4);            // [256][512] f32
constexpr size_t MO_COEF   = aln(MO_EXV    + size_t(B_) * D_ * 4);            // [256][8]   f32
constexpr size_t MO_CTXB   = aln(MO_COEF   + size_t(B_) * 8  * 4);            // [256][512] bf16
constexpr size_t MO_ABUF   = aln(MO_CTXB   + size_t(B_) * D_ * 2);            // [256][512] f32 (a, then ff)
constexpr size_t MO_H1F    = aln(MO_ABUF   + size_t(B_) * D_ * 4);            // [256][512] f32
constexpr size_t MO_H1B    = aln(MO_H1F    + size_t(B_) * D_ * 4);            // [256][512] bf16
constexpr size_t MO_GBUF   = aln(MO_H1B    + size_t(B_) * D_ * 2);            // [256][2048] bf16
constexpr size_t MO_SCORES = aln(MO_GBUF   + size_t(B_) * DFF_ * 2);          // [256][8][50] f32
constexpr size_t MO_LASTP  = aln(MO_SCORES + size_t(B_) * NH_ * 50 * 4);      // [256][2] f32
constexpr size_t MODE_STRIDE = aln(MO_LASTP + size_t(B_) * 2 * 4);

DEV unsigned short f2bf(float f) {
  union { float f; unsigned u; } c; c.f = f;
  unsigned u = c.u;
  return (unsigned short)((u + 0x7FFFu + ((u >> 16) & 1u)) >> 16);
}

DEV float wave_sum(float v) {
  v += __shfl_xor(v, 16, 32);
  v += __shfl_xor(v, 8, 32);
  v += __shfl_xor(v, 4, 32);
  v += __shfl_xor(v, 2, 32);
  v += __shfl_xor(v, 1, 32);
  return v;
}

DEV v8f wmma_bf16(const v16bf& a, const v16bf& b, const v8f& c) {
  return __builtin_amdgcn_wmma_f32_16x16x32_bf16(false, a, false, b, (short)0, c,
                                                 false, false);
}

// C[256xN] = A[256xK](bf16,row-major) * W[KxN] (given as WT[N][K] bf16) + bias
// 2x2 register blocking: each wave owns a 32x32 output supertile.
// WMMA 16x16x32 bf16 -> f32. ISA layouts (CDNA5 ISA):
//   A 16x32: lane=(r=M%16, h=K-half), VGPR j packs K = (j<4 ? 2j : 2j+8)+8h, +1
//   B 32x16: lane=(n, h),            VGPR j packs K = (j<4 ? 2j : 2j+8)+8h, +1
//   C 16x16: VGPR v -> M = v+8h, N = lane&15
DEV void gemm_bf16(const unsigned short* __restrict__ A,
                   const unsigned short* __restrict__ WT,
                   const float* __restrict__ bias,
                   float* __restrict__ Cf, unsigned short* __restrict__ Cb,
                   int K, int N, bool relu, int tid) {
  const int lane = tid & 31, wave = tid >> 5;
  const int r = lane & 15, h = lane >> 4;
  const int nsN = N >> 5;                       // 32-wide col supertiles
  const int nsup = 8 * nsN;                     // 8 row supertiles (256 rows)
  for (int t = wave; t < nsup; t += (BLOCK >> 5)) {
    const int ms = t & 7, ns = t >> 3;
    const unsigned short* Ap0 = A  + size_t(ms * 32 + r) * K;
    const unsigned short* Ap1 = Ap0 + size_t(16) * K;
    const unsigned short* Bp0 = WT + size_t(ns * 32 + r) * K;
    const unsigned short* Bp1 = Bp0 + size_t(16) * K;
    v8f acc00 = {}, acc01 = {}, acc10 = {}, acc11 = {};
    for (int kk = 0; kk < K; kk += 32) {
      union { v16bf v; unsigned u[8]; } a0, a1, b0, b1;
#pragma unroll
      for (int j = 0; j < 8; ++j) {
        const int ka = kk + ((j < 4) ? 2 * j : 2 * j + 8) + 8 * h;
        const int kb = kk + ((j < 4) ? 2 * j : 2 * j + 8) + 8 * h;
        a0.u[j] = *(const unsigned*)(Ap0 + ka);
        a1.u[j] = *(const unsigned*)(Ap1 + ka);
        b0.u[j] = *(const unsigned*)(Bp0 + kb);
        b1.u[j] = *(const unsigned*)(Bp1 + kb);
      }
      acc00 = wmma_bf16(a0.v, b0.v, acc00);
      acc01 = wmma_bf16(a0.v, b1.v, acc01);
      acc10 = wmma_bf16(a1.v, b0.v, acc10);
      acc11 = wmma_bf16(a1.v, b1.v, acc11);
    }
    const int col0 = ns * 32 + r, col1 = col0 + 16;
    const float bv0 = bias ? bias[col0] : 0.f;
    const float bv1 = bias ? bias[col1] : 0.f;
#pragma unroll
    for (int v = 0; v < 8; ++v) {
      const int row0 = ms * 32 + v + 8 * h;
      const int row1 = row0 + 16;
      float x00 = acc00[v] + bv0, x01 = acc01[v] + bv1;
      float x10 = acc10[v] + bv0, x11 = acc11[v] + bv1;
      if (relu) {
        x00 = fmaxf(x00, 0.f); x01 = fmaxf(x01, 0.f);
        x10 = fmaxf(x10, 0.f); x11 = fmaxf(x11, 0.f);
      }
      if (Cf) {
        Cf[size_t(row0) * N + col0] = x00; Cf[size_t(row0) * N + col1] = x01;
        Cf[size_t(row1) * N + col0] = x10; Cf[size_t(row1) * N + col1] = x11;
      }
      if (Cb) {
        Cb[size_t(row0) * N + col0] = f2bf(x00); Cb[size_t(row0) * N + col1] = f2bf(x01);
        Cb[size_t(row1) * N + col0] = f2bf(x10); Cb[size_t(row1) * N + col1] = f2bf(x11);
      }
    }
  }
}

// ---------------- precompute kernels ----------------

// dst[N][K] bf16 = transpose of src[K][N] f32
__global__ void k_tcvt(const float* __restrict__ src, unsigned short* __restrict__ dst,
                       int K, int N) {
  int idx = blockIdx.x * blockDim.x + threadIdx.x;
  if (idx >= K * N) return;
  int n = idx / K, k = idx - n * K;
  dst[idx] = f2bf(src[size_t(k) * N + n]);
}

// wexT[j][c]: row sums of W_qkv rows 32..63 grouped by (gx,gy,lx,ly,dx,dy,t),
// stored transposed: 8 contiguous floats per output column j.
__global__ void k_wexred(const float* __restrict__ Wqkv, float* __restrict__ wexT) {
  int j = blockIdx.x * blockDim.x + threadIdx.x;
  if (j >= 1536) return;
  float* o = wexT + j * 8;
  o[0] = Wqkv[32*1536+j] + Wqkv[34*1536+j] + Wqkv[36*1536+j] + Wqkv[38*1536+j];
  o[1] = Wqkv[33*1536+j] + Wqkv[35*1536+j] + Wqkv[37*1536+j] + Wqkv[39*1536+j];
  o[2] = Wqkv[40*1536+j] + Wqkv[42*1536+j] + Wqkv[44*1536+j] + Wqkv[46*1536+j];
  o[3] = Wqkv[41*1536+j] + Wqkv[43*1536+j] + Wqkv[45*1536+j] + Wqkv[47*1536+j];
  o[4] = Wqkv[48*1536+j] + Wqkv[50*1536+j] + Wqkv[52*1536+j] + Wqkv[54*1536+j];
  o[5] = Wqkv[49*1536+j] + Wqkv[51*1536+j] + Wqkv[53*1536+j] + Wqkv[55*1536+j];
  float s = 0.f;
  for (int r = 56; r < 64; ++r) s += Wqkv[r*1536+j];
  o[6] = s;
  o[7] = 0.f;
}

// WqT[d][e] = W_qkv[e][d] ; WvT[d][e] = W_qkv[e][1024+d]  (emb blocks, e<32)
__global__ void k_embT(const float* __restrict__ Wqkv, float* __restrict__ wqT,
                       float* __restrict__ wvT) {
  int idx = blockIdx.x * blockDim.x + threadIdx.x;
  if (idx >= D_ * EMB_) return;
  int d = idx >> 5, e = idx & 31;
  wqT[idx] = Wqkv[e * 1536 + d];
  wvT[idx] = Wqkv[e * 1536 + 1024 + d];
}

// featc[b][1536] = b_qkv + feat[b] @ W_qkv[64:512]
__global__ void k_featconst(const float* __restrict__ feat, const float* __restrict__ Wqkv,
                            const float* __restrict__ bqkv, float* __restrict__ featc) {
  int idx = blockIdx.x * blockDim.x + threadIdx.x;
  if (idx >= B_ * 1536) return;
  int b = idx / 1536, j = idx - b * 1536;
  float s = bqkv[j];
  const float* fb = feat + size_t(b) * INC_;
  for (int e = 0; e < INC_; ++e) s += fb[e] * Wqkv[size_t(64 + e) * 1536 + j];
  featc[idx] = s;
}

// emb_obs[l][b][e] = relu(obs[l][b]@W_emb + b_emb)
__global__ void k_embobs(const float* __restrict__ obs, const float* __restrict__ Wemb,
                         const float* __restrict__ bemb, float* __restrict__ dst) {
  int idx = blockIdx.x * blockDim.x + threadIdx.x;
  if (idx >= OBS_ * B_ * EMB_) return;
  int e = idx & 31, lb = idx >> 5;           // lb = l*B + b
  float v = obs[lb * 2 + 0] * Wemb[e] + obs[lb * 2 + 1] * Wemb[32 + e] + bemb[e];
  dst[idx] = fmaxf(v, 0.f);
}

// Wfg = W_fus @ W_out ; bfg = b_fus @ W_out + b_out
__global__ void k_wfg(const float* __restrict__ Wfus, const float* __restrict__ bfus,
                      const float* __restrict__ Wout, const float* __restrict__ bout,
                      float* __restrict__ wfg, float* __restrict__ bfg) {
  int idx = blockIdx.x * blockDim.x + threadIdx.x;
  if (idx < FUS_ * 2) {
    int i = idx >> 1, j = idx & 1;
    float s = 0.f;
    for (int h = 0; h < 448; ++h) s += Wfus[size_t(i) * 448 + h] * Wout[h * 2 + j];
    wfg[idx] = s;
  } else if (idx < FUS_ * 2 + 2) {
    int j = idx - FUS_ * 2;
    float s = bout[j];
    for (int h = 0; h < 448; ++h) s += bfus[h] * Wout[h * 2 + j];
    bfg[j] = s;
  }
}

// featfg[b][j] = feat[b] @ Wfg[519:967]
__global__ void k_featfg(const float* __restrict__ feat, const float* __restrict__ wfg,
                         float* __restrict__ featfg) {
  int idx = blockIdx.x * blockDim.x + threadIdx.x;
  if (idx >= B_ * 2) return;
  int b = idx >> 1, j = idx & 1;
  float s = 0.f;
  const float* fb = feat + size_t(b) * INC_;
  for (int e = 0; e < INC_; ++e) s += fb[e] * wfg[(519 + e) * 2 + j];
  featfg[idx] = s;
}

// ---------------- main persistent rollout ----------------

__global__ __launch_bounds__(BLOCK) void k_main(
    const float* __restrict__ feat, const float* __restrict__ loc_in,
    const float* __restrict__ obs,  const float* __restrict__ Wemb,
    const float* __restrict__ bemb, const float* __restrict__ Wqkv,
    const float* __restrict__ b_o,  const float* __restrict__ b_ff1,
    const float* __restrict__ b_ff2,
    const float* __restrict__ ln1g, const float* __restrict__ ln1b,
    const float* __restrict__ ln2g, const float* __restrict__ ln2b,
    char* __restrict__ ws, float* __restrict__ out) {
  const int m = blockIdx.x;
  const int tid = threadIdx.x;
  const int lane = tid & 31, wave = tid >> 5;

  const unsigned short* WoT   = (const unsigned short*)(ws + OFF_WOT);
  const unsigned short* Wff1T = (const unsigned short*)(ws + OFF_WFF1T);
  const unsigned short* Wff2T = (const unsigned short*)(ws + OFF_WFF2T);
  const float* featc  = (const float*)(ws + OFF_FEATC);
  const float* wexT   = (const float*)(ws + OFF_WEXT);
  const float* wqT    = (const float*)(ws + OFF_WQT);
  const float* wvT    = (const float*)(ws + OFF_WVT);
  const float* embobs = (const float*)(ws + OFF_EMBOBS);
  const float* wfg    = (const float*)(ws + OFF_WFG);
  const float* bfg    = (const float*)(ws + OFF_BFG);
  const float* featfg = (const float*)(ws + OFF_FEATFG);

  char* mb = ws + OFF_MODE + size_t(m) * MODE_STRIDE;
  float* emb30 = (float*)(mb + MO_EMB30);
  float* qbuf  = (float*)(mb + MO_Q);
  float* exv   = (float*)(mb + MO_EXV);
  float* coef  = (float*)(mb + MO_COEF);
  unsigned short* ctxb = (unsigned short*)(mb + MO_CTXB);
  float* abuf  = (float*)(mb + MO_ABUF);
  float* h1f   = (float*)(mb + MO_H1F);
  unsigned short* h1b  = (unsigned short*)(mb + MO_H1B);
  unsigned short* gbuf = (unsigned short*)(mb + MO_GBUF);
  float* scores = (float*)(mb + MO_SCORES);
  float* lastp  = (float*)(mb + MO_LASTP);

  for (int b = tid; b < B_; b += BLOCK) {
    lastp[b * 2 + 0] = obs[(19 * B_ + b) * 2 + 0];
    lastp[b * 2 + 1] = obs[(19 * B_ + b) * 2 + 1];
  }
  __syncthreads();

  for (int fi = OBS_; fi < OBS_ + HOR_; ++fi) {
    // ---- A0: extras coefficients [gx,gy,lx,ly,dx,dy,t] ----
    for (int b = tid; b < B_; b += BLOCK) {
      float lx = lastp[b * 2], ly = lastp[b * 2 + 1];
      float gx = loc_in[(b * MODES + m) * 2], gy = loc_in[(b * MODES + m) * 2 + 1];
      float* cf = coef + b * 8;
      cf[0] = gx; cf[1] = gy; cf[2] = lx; cf[3] = ly;
      cf[4] = gx - lx; cf[5] = gy - ly; cf[6] = (float)fi;
    }
    __syncthreads();

    const float* embprev = (fi - 1 < OBS_)
        ? (embobs + size_t(fi - 1) * B_ * EMB_)
        : (emb30  + size_t(fi - 1 - OBS_) * B_ * EMB_);

    // ---- A1: q = emb_prev@Wq_emb + featq + exq ;  exv = extras@Wv_ex ----
    for (int idx = tid; idx < B_ * D_; idx += BLOCK) {
      int b = idx >> 9, d = idx & (D_ - 1);
      const float* cf = coef + b * 8;
      const float* wq8 = wexT + d * 8;
      const float* wv8 = wexT + (1024 + d) * 8;
      float eq = 0.f, ev = 0.f;
#pragma unroll
      for (int c = 0; c < 7; ++c) {
        eq += cf[c] * wq8[c];
        ev += cf[c] * wv8[c];
      }
      exv[idx] = ev;
      const float* ep = embprev + b * EMB_;
      const float* wq = wqT + d * EMB_;
      float s = featc[size_t(b) * 1536 + d] + eq;
#pragma unroll
      for (int e = 0; e < EMB_; ++e) s += ep[e] * wq[e];
      qbuf[idx] = s;
    }
    __syncthreads();

    // ---- B: attention per (b,h). K/V never materialized:
    //   scores[l] = emb[l]·qW (l-constant part cancels in softmax)
    //   ctx       = (Σ attn·emb) @ Wv_emb + featv + exv
    for (int idx = tid; idx < B_ * NH_; idx += BLOCK) {
      int b = idx >> 3, h = idx & 7;
      const float* qb = qbuf + size_t(b) * D_ + h * 64;
      float qW[32];
#pragma unroll
      for (int e = 0; e < 32; ++e) {
        const float* wr = Wqkv + e * 1536 + 512 + h * 64;   // contiguous over d
        float s = 0.f;
        for (int d = 0; d < 64; ++d) s += qb[d] * wr[d];
        qW[e] = s;
      }
      float* sc = scores + size_t(b * NH_ + h) * 50;
      float mx = -1e30f;
      for (int l = 0; l < fi; ++l) {
        const float* el = (l < OBS_) ? (embobs + size_t(l * B_ + b) * EMB_)
                                     : (emb30 + size_t((l - OBS_) * B_ + b) * EMB_);
        float s = 0.f;
#pragma unroll
        for (int e = 0; e < 32; ++e) s += el[e] * qW[e];
        s *= 0.125f;                           // 1/sqrt(HD)
        sc[l] = s;
        mx = fmaxf(mx, s);
      }
      float ebar[32];
#pragma unroll
      for (int e = 0; e < 32; ++e) ebar[e] = 0.f;
      float sum = 0.f;
      for (int l = 0; l < fi; ++l) {
        const float* el = (l < OBS_) ? (embobs + size_t(l * B_ + b) * EMB_)
                                     : (emb30 + size_t((l - OBS_) * B_ + b) * EMB_);
        float p = __expf(sc[l] - mx);
        sum += p;
#pragma unroll
        for (int e = 0; e < 32; ++e) ebar[e] += p * el[e];
      }
      float inv = 1.f / sum;
      for (int d = 0; d < 64; ++d) {
        const float* wv = wvT + (h * 64 + d) * EMB_;        // contiguous over e
        float s = 0.f;
#pragma unroll
        for (int e = 0; e < 32; ++e) s += ebar[e] * wv[e];
        s = s * inv + featc[size_t(b) * 1536 + 1024 + h * 64 + d]
                    + exv[size_t(b) * D_ + h * 64 + d];
        ctxb[size_t(b) * D_ + h * 64 + d] = f2bf(s);
      }
    }
    __syncthreads();

    // ---- C: a = ctx @ W_o + b_o  (WMMA) ----
    gemm_bf16(ctxb, WoT, b_o, abuf, nullptr, D_, D_, false, tid);
    __syncthreads();

    // ---- C2: h1 = LN(x_last + a), x_last = [emb_prev | extras | feat] ----
    for (int b = wave; b < B_; b += (BLOCK >> 5)) {
      float xv[16], s1 = 0.f;
      const float* cf = coef + b * 8;
#pragma unroll
      for (int i = 0; i < 16; ++i) {
        int d = lane + i * 32;
        float x;
        if (d < 32) x = embprev[b * EMB_ + d];
        else if (d < 64) {
          int e = d - 32;
          x = (e < 8) ? cf[e & 1] : (e < 16) ? cf[2 + (e & 1)]
            : (e < 24) ? cf[4 + (e & 1)] : cf[6];
        } else x = feat[size_t(b) * INC_ + d - 64];
        x += abuf[size_t(b) * D_ + d];
        xv[i] = x; s1 += x;
      }
      float mean = wave_sum(s1) * (1.f / 512.f);
      float s2 = 0.f;
#pragma unroll
      for (int i = 0; i < 16; ++i) { float c = xv[i] - mean; s2 += c * c; }
      float rstd = rsqrtf(wave_sum(s2) * (1.f / 512.f) + 1e-5f);
#pragma unroll
      for (int i = 0; i < 16; ++i) {
        int d = lane + i * 32;
        float y = (xv[i] - mean) * rstd * ln1g[d] + ln1b[d];
        h1f[size_t(b) * D_ + d] = y;
        h1b[size_t(b) * D_ + d] = f2bf(y);
      }
    }
    __syncthreads();

    // ---- D: g = relu(h1 @ W_ff1 + b_ff1)  (WMMA) ----
    gemm_bf16(h1b, Wff1T, b_ff1, nullptr, gbuf, D_, DFF_, true, tid);
    __syncthreads();

    // ---- E: ff = g @ W_ff2 + b_ff2  (WMMA) ----
    gemm_bf16(gbuf, Wff2T, b_ff2, abuf, nullptr, DFF_, D_, false, tid);
    __syncthreads();

    // ---- E2: h2 = LN(h1 + ff) -> qbuf (q is dead) ----
    for (int b = wave; b < B_; b += (BLOCK >> 5)) {
      float xv[16], s1 = 0.f;
#pragma unroll
      for (int i = 0; i < 16; ++i) {
        int d = lane + i * 32;
        float x = h1f[size_t(b) * D_ + d] + abuf[size_t(b) * D_ + d];
        xv[i] = x; s1 += x;
      }
      float mean = wave_sum(s1) * (1.f / 512.f);
      float s2 = 0.f;
#pragma unroll
      for (int i = 0; i < 16; ++i) { float c = xv[i] - mean; s2 += c * c; }
      float rstd = rsqrtf(wave_sum(s2) * (1.f / 512.f) + 1e-5f);
#pragma unroll
      for (int i = 0; i < 16; ++i) {
        int d = lane + i * 32;
        qbuf[size_t(b) * D_ + d] = (xv[i] - mean) * rstd * ln2g[d] + ln2b[d];
      }
    }
    __syncthreads();

    // ---- F: out = fus @ Wfg + bfg ; update pos/emb/output ----
    for (int b = tid; b < B_; b += BLOCK) {
      const float* h2 = qbuf + size_t(b) * D_;
      float o0 = bfg[0] + featfg[b * 2 + 0];
      float o1 = bfg[1] + featfg[b * 2 + 1];
      for (int d = 0; d < D_; ++d) {
        float v = h2[d];
        o0 += v * wfg[d * 2 + 0];
        o1 += v * wfg[d * 2 + 1];
      }
      const float* cf = coef + b * 8;
      const float s7[7] = { cf[2], cf[3], cf[0], cf[1], cf[4], cf[5], cf[6] };
#pragma unroll
      for (int c = 0; c < 7; ++c) {
        o0 += s7[c] * wfg[(512 + c) * 2 + 0];
        o1 += s7[c] * wfg[(512 + c) * 2 + 1];
      }
      lastp[b * 2 + 0] = o0;
      lastp[b * 2 + 1] = o1;
      const int step = fi - OBS_;
      out[((size_t(b) * MODES + m) * HOR_ + step) * 2 + 0] = o0;
      out[((size_t(b) * MODES + m) * HOR_ + step) * 2 + 1] = o1;
      float* er = emb30 + size_t(step * B_ + b) * EMB_;
#pragma unroll
      for (int e = 0; e < EMB_; ++e) {
        float v = o0 * Wemb[e] + o1 * Wemb[32 + e] + bemb[e];
        er[e] = fmaxf(v, 0.f);
      }
    }
    __syncthreads();
  }
}

// ---------------------------------------------------------------------------

extern "C" void kernel_launch(void* const* d_in, const int* in_sizes, int n_in,
                              void* d_out, int out_size, void* d_ws, size_t ws_size,
                              hipStream_t stream) {
  const float* feat    = (const float*)d_in[0];   // (B,1,INC)
  const float* loc_in  = (const float*)d_in[1];   // (B,M,2)
  const float* obs     = (const float*)d_in[2];   // (OBS,B,2)
  const float* W_emb   = (const float*)d_in[3];
  const float* b_emb   = (const float*)d_in[4];
  const float* W_qkv   = (const float*)d_in[5];
  const float* b_qkv   = (const float*)d_in[6];
  const float* W_o     = (const float*)d_in[7];
  const float* b_o     = (const float*)d_in[8];
  const float* W_ff1   = (const float*)d_in[9];
  const float* b_ff1   = (const float*)d_in[10];
  const float* W_ff2   = (const float*)d_in[11];
  const float* b_ff2   = (const float*)d_in[12];
  const float* ln1_g   = (const float*)d_in[13];
  const float* ln1_b   = (const float*)d_in[14];
  const float* ln2_g   = (const float*)d_in[15];
  const float* ln2_b   = (const float*)d_in[16];
  const float* W_fus   = (const float*)d_in[17];
  const float* b_fus   = (const float*)d_in[18];
  const float* W_out   = (const float*)d_in[19];
  const float* b_out   = (const float*)d_in[20];
  (void)in_sizes; (void)n_in; (void)out_size; (void)ws_size;

  char* ws = (char*)d_ws;
  unsigned short* WoT   = (unsigned short*)(ws + OFF_WOT);
  unsigned short* Wff1T = (unsigned short*)(ws + OFF_WFF1T);
  unsigned short* Wff2T = (unsigned short*)(ws + OFF_WFF2T);
  float* featc  = (float*)(ws + OFF_FEATC);
  float* wexT   = (float*)(ws + OFF_WEXT);
  float* wqT    = (float*)(ws + OFF_WQT);
  float* wvT    = (float*)(ws + OFF_WVT);
  float* embobs = (float*)(ws + OFF_EMBOBS);
  float* wfg    = (float*)(ws + OFF_WFG);
  float* bfg    = (float*)(ws + OFF_BFG);
  float* featfg = (float*)(ws + OFF_FEATFG);

  // weight transpose + bf16 conversion
  k_tcvt<<<(D_ * D_ + 255) / 256, 256, 0, stream>>>(W_o, WoT, D_, D_);
  k_tcvt<<<(D_ * DFF_ + 255) / 256, 256, 0, stream>>>(W_ff1, Wff1T, D_, DFF_);
  k_tcvt<<<(DFF_ * D_ + 255) / 256, 256, 0, stream>>>(W_ff2, Wff2T, DFF_, D_);
  // algebraic precomputes
  k_wexred<<<(1536 + 255) / 256, 256, 0, stream>>>(W_qkv, wexT);
  k_embT<<<(D_ * EMB_ + 255) / 256, 256, 0, stream>>>(W_qkv, wqT, wvT);
  k_featconst<<<(B_ * 1536 + 255) / 256, 256, 0, stream>>>(feat, W_qkv, b_qkv, featc);
  k_embobs<<<(OBS_ * B_ * EMB_ + 255) / 256, 256, 0, stream>>>(obs, W_emb, b_emb, embobs);
  k_wfg<<<(FUS_ * 2 + 2 + 255) / 256, 256, 0, stream>>>(W_fus, b_fus, W_out, b_out, wfg, bfg);
  k_featfg<<<(B_ * 2 + 255) / 256, 256, 0, stream>>>(feat, wfg, featfg);

  // persistent rollout: one workgroup per mode
  k_main<<<MODES, BLOCK, 0, stream>>>(feat, loc_in, obs, W_emb, b_emb, W_qkv,
                                      b_o, b_ff1, b_ff2, ln1_g, ln1_b, ln2_g, ln2_b,
                                      ws, (float*)d_out);
}
